// Seq2SeqDaVinciNet_41979010351513
// MI455X (gfx1250) — matrix operand order repaired and checked by reading, not compiled, not run on hardware
//
#include <hip/hip_runtime.h>

// ---------------- model constants ----------------
#define BB   4096
#define SS   64
#define DD   9
#define HH   128
#define GG   512      // 4*H
#define HZN  32
#define MB   16       // batch rows per workgroup (one WMMA M-tile)
#define NTH  256      // 8 wave32
#define DT_  0.1f

typedef __attribute__((ext_vector_type(16))) _Float16 v16h;
typedef __attribute__((ext_vector_type(8)))  float    v8f;
typedef _Float16 f16;

// ---------------- workspace layout (bytes) ----------------
// fp16 weight copies (L2-resident, ~0.57MB) + fp32 bias sums + recurrent dumps
constexpr size_t SZ_WIHE  = (size_t)GG * 32 * 2;        // W_ih_enc padded K=9->32
constexpr size_t SZ_WHHE  = (size_t)GG * HH * 2;
constexpr size_t SZ_WE    = (size_t)SS * 2 * HH * 2;
constexpr size_t SZ_WD    = (size_t)HH * 2 * HH * 2;
constexpr size_t SZ_VD    = (size_t)HH * HH * 2;
constexpr size_t SZ_WPD   = (size_t)GG * 32 * 2;        // W_ih_dec[:, :9] padded
constexpr size_t SZ_WCD   = (size_t)GG * HH * 2;        // W_ih_dec[:, 9:137]
constexpr size_t SZ_WHHD  = (size_t)GG * HH * 2;
constexpr size_t SZ_BIAS  = (size_t)GG * 4;
constexpr size_t SZ_HENC  = (size_t)BB * HH * 4;        // final h (f32)
constexpr size_t SZ_CENC  = (size_t)BB * HH * 4;        // final c (f32)
constexpr size_t SZ_ENCH  = (size_t)BB * SS * HH * 2;   // all encoder h_t (f16)
constexpr size_t SZ_ATTNE = (size_t)BB * SS * HH * 2;   // enc_hidden @ V_d^T (f16)

constexpr size_t OFF_WIHE  = 0;
constexpr size_t OFF_WHHE  = OFF_WIHE  + SZ_WIHE;
constexpr size_t OFF_WE    = OFF_WHHE  + SZ_WHHE;
constexpr size_t OFF_WD    = OFF_WE    + SZ_WE;
constexpr size_t OFF_VD    = OFF_WD    + SZ_WD;
constexpr size_t OFF_WPD   = OFF_VD    + SZ_VD;
constexpr size_t OFF_WCD   = OFF_WPD   + SZ_WPD;
constexpr size_t OFF_WHHD  = OFF_WCD   + SZ_WCD;
constexpr size_t OFF_BIASE = OFF_WHHD  + SZ_WHHD;
constexpr size_t OFF_BIASD = OFF_BIASE + SZ_BIAS;
constexpr size_t OFF_HENC  = OFF_BIASD + SZ_BIAS;
constexpr size_t OFF_CENC  = OFF_HENC  + SZ_HENC;
constexpr size_t OFF_ENCH  = OFF_CENC  + SZ_CENC;
constexpr size_t OFF_ATTNE = OFF_ENCH  + SZ_ENCH;

constexpr int PREP_TOTAL = GG*32 + GG*HH + SS*2*HH + HH*2*HH + HH*HH
                         + GG*32 + GG*HH + GG*HH + GG + GG;

// ---------------- WMMA fragment helpers (A/B: contiguous k halves; C/D: ISA layout) ----------------
union H16U { v16h v; unsigned int u[8]; };

// A: 16xK row-major f16 tile at (row0, col k0); ld = K stride.
// lanes 0-15: M=lane, K = 0..15; lanes 16-31: M=lane-16, K = 16..31 (contiguous)
__device__ __forceinline__ v16h load_a_frag(const f16* base, int ld) {
  int lane = threadIdx.x & 31;
  int m    = lane & 15;
  int kh   = (lane >> 4) << 4;           // 0 or 16
  const f16* r = base + m * ld + kh;
  H16U h;
#pragma unroll
  for (int i = 0; i < 4; ++i) h.u[i]   = *(const unsigned int*)(r + 2*i);
#pragma unroll
  for (int i = 0; i < 4; ++i) h.u[4+i] = *(const unsigned int*)(r + 8 + 2*i);
  return h.v;
}

// B: K x N fragment taken from W stored row-major (Ntotal x Ktotal), i.e. B = W^T.
// lanes 0-15: N=lane, K=k0..k0+15; lanes 16-31: N=lane-16, K=k0+16..k0+31 (contiguous!)
__device__ __forceinline__ v16h load_b_frag(const f16* W, int ldk, int n0, int k0) {
  int lane = threadIdx.x & 31;
  int n  = n0 + (lane & 15);
  int kb = k0 + ((lane >> 4) << 4);
  const f16* r = W + (size_t)n * ldk + kb;
  H16U h;
#pragma unroll
  for (int i = 0; i < 8; ++i) h.u[i] = *(const unsigned int*)(r + 2*i);
  return h.v;
}

// C/D f32: VGPR i holds M=i (lanes 0-15) / M=8+i (lanes 16-31); N = lane&15
__device__ __forceinline__ void store_d_f32(float* dst, int ld, int n0, const v8f& c) {
  int lane = threadIdx.x & 31;
  int n  = n0 + (lane & 15);
  int mb = (lane >> 4) << 3;
#pragma unroll
  for (int i = 0; i < 8; ++i) dst[(mb + i) * ld + n] = c[i];
}
__device__ __forceinline__ void store_d_f16(f16* dst, int ld, int n0, const v8f& c) {
  int lane = threadIdx.x & 31;
  int n  = n0 + (lane & 15);
  int mb = (lane >> 4) << 3;
#pragma unroll
  for (int i = 0; i < 8; ++i) dst[(mb + i) * ld + n] = (f16)c[i];
}

__device__ __forceinline__ v8f wmma16(const v16h& a, const v16h& b, const v8f& c) {
  return __builtin_amdgcn_wmma_f32_16x16x32_f16(false, a, false, b, (short)0, c, false, false);
}

__device__ __forceinline__ float sig_(float x)  { return 1.f / (1.f + __expf(-x)); }
__device__ __forceinline__ float tanh_(float x) { return 2.f * sig_(2.f * x) - 1.f; }

// ---------------- kernel 1: fp32 -> fp16 weight prep + bias sums ----------------
__global__ void prep_kernel(const float* WihE, const float* WhhE,
                            const float* bihE, const float* bhhE,
                            const float* We, const float* Wd, const float* Vd,
                            const float* WihD, const float* WhhD,
                            const float* bihD, const float* bhhD, char* ws) {
  int i = blockIdx.x * blockDim.x + threadIdx.x;
  f16*   pWihE = (f16*)(ws + OFF_WIHE);
  f16*   pWhhE = (f16*)(ws + OFF_WHHE);
  f16*   pWe   = (f16*)(ws + OFF_WE);
  f16*   pWd   = (f16*)(ws + OFF_WD);
  f16*   pVd   = (f16*)(ws + OFF_VD);
  f16*   pWpD  = (f16*)(ws + OFF_WPD);
  f16*   pWcD  = (f16*)(ws + OFF_WCD);
  f16*   pWhhD = (f16*)(ws + OFF_WHHD);
  float* pBE   = (float*)(ws + OFF_BIASE);
  float* pBD   = (float*)(ws + OFF_BIASD);

  if (i < GG*32) { int r = i >> 5, c = i & 31;
    pWihE[i] = (f16)(c < DD ? WihE[r*DD + c] : 0.f); return; } i -= GG*32;
  if (i < GG*HH) { pWhhE[i] = (f16)WhhE[i]; return; } i -= GG*HH;
  if (i < SS*2*HH) { pWe[i] = (f16)We[i]; return; } i -= SS*2*HH;
  if (i < HH*2*HH) { pWd[i] = (f16)Wd[i]; return; } i -= HH*2*HH;
  if (i < HH*HH)   { pVd[i] = (f16)Vd[i]; return; } i -= HH*HH;
  if (i < GG*32) { int r = i >> 5, c = i & 31;
    pWpD[i] = (f16)(c < DD ? WihD[r*(DD+HH) + c] : 0.f); return; } i -= GG*32;
  if (i < GG*HH) { int r = i >> 7, c = i & (HH-1);
    pWcD[i] = (f16)WihD[r*(DD+HH) + DD + c]; return; } i -= GG*HH;
  if (i < GG*HH) { pWhhD[i] = (f16)WhhD[i]; return; } i -= GG*HH;
  if (i < GG) { pBE[i] = bihE[i] + bhhE[i]; return; } i -= GG;
  if (i < GG) { pBD[i] = bihD[i] + bhhD[i]; }
}

// ---------------- kernel 2: encoder (input attention + LSTM, S=64 steps) ----------------
__global__ __launch_bounds__(NTH)
void encoder_kernel(const float* __restrict__ x, const float* __restrict__ Ve,
                    const float* __restrict__ ve, char* ws) {
  __shared__ f16   sAttnIn[MB][DD][SS];   // 18 KB  V_e * x^i (precomputed per tile)
  __shared__ f16   sHC[MB][2*HH];         //  8 KB  [h|c] fp16 (WMMA A source)
  __shared__ float sC[MB][HH];            //  8 KB  fp32 cell state
  __shared__ float sAttnH[MB][SS];        //  4 KB
  __shared__ f16   sGates[MB][GG];        // 16 KB
  __shared__ float sAlpha[MB][16];        //  1 KB
  __shared__ f16   sXpad[MB][32];         //  1 KB  alpha*x_t padded K=9->32

  const f16*   wWe  = (const f16*)(ws + OFF_WE);
  const f16*   wWih = (const f16*)(ws + OFF_WIHE);
  const f16*   wWhh = (const f16*)(ws + OFF_WHHE);
  const float* bias = (const float*)(ws + OFF_BIASE);
  f16*   encH = (f16*)(ws + OFF_ENCH);
  float* hEnd = (float*)(ws + OFF_HENC);
  float* cEnd = (float*)(ws + OFF_CENC);

  const int b0   = blockIdx.x * MB;
  const int tid  = threadIdx.x;
  const int wave = tid >> 5;

  for (int i = tid; i < MB*2*HH; i += NTH) (&sHC[0][0])[i] = (f16)0.f;
  for (int i = tid; i < MB*HH;   i += NTH) (&sC[0][0])[i]  = 0.f;

  // attn_input[m][d][t] = sum_s x[b,s,d] * Ve[t,s]
  for (int idx = tid; idx < MB*DD*SS; idx += NTH) {
    int t = idx % SS, dd = (idx / SS) % DD, m = idx / (SS*DD);
    const float* xb = x + ((size_t)(b0+m))*SS*DD + dd;
    const float* vt = Ve + t*SS;
    float acc = 0.f;
    for (int s = 0; s < SS; ++s) acc += xb[s*DD] * vt[s];
    sAttnIn[m][dd][t] = (f16)acc;
  }
  __syncthreads();

  for (int t = 0; t < SS; ++t) {
    // (1) attn_hidden = [h|c] @ W_e^T : M=16, N=64, K=256 (waves 0..3)
    if (wave < 4) {
      int n0 = wave * 16;
      v8f acc = {};
#pragma unroll
      for (int k0 = 0; k0 < 2*HH; k0 += 32)
        acc = wmma16(load_a_frag(&sHC[0][0] + k0, 2*HH),
                     load_b_frag(wWe, 2*HH, n0, k0), acc);
      store_d_f32(&sAttnH[0][0], SS, n0, acc);
    }
    __syncthreads();

    // (2) e[m][d] = sum_t v_e[t]*tanh(attn_h + attn_in); softmax over D; xpad
    for (int idx = tid; idx < MB*DD; idx += NTH) {
      int dd = idx % DD, m = idx / DD;
      float acc = 0.f;
#pragma unroll 4
      for (int s = 0; s < SS; ++s)
        acc += ve[s] * tanh_(sAttnH[m][s] + (float)sAttnIn[m][dd][s]);
      sAlpha[m][dd] = acc;
    }
    __syncthreads();
    if (tid < MB) {
      int m = tid; float mx = -1e30f;
      for (int dd = 0; dd < DD; ++dd) mx = fmaxf(mx, sAlpha[m][dd]);
      float sum = 0.f, ex[DD];
      for (int dd = 0; dd < DD; ++dd) { ex[dd] = __expf(sAlpha[m][dd]-mx); sum += ex[dd]; }
      float inv = 1.f / sum;
      for (int dd = 0; dd < DD; ++dd) sAlpha[m][dd] = ex[dd] * inv;
    }
    __syncthreads();
    for (int idx = tid; idx < MB*32; idx += NTH) {
      int k = idx & 31, m = idx >> 5;
      float v = 0.f;
      if (k < DD) v = sAlpha[m][k] * x[((size_t)(b0+m))*SS*DD + t*DD + k];
      sXpad[m][k] = (f16)v;
    }
    __syncthreads();

    // (3) gates = xpad@Wih^T + h@Whh^T : M=16, N=512; 4 N-tiles/wave, 5 WMMA each
    {
      v16h ax = load_a_frag(&sXpad[0][0], 32);
      v16h ah[4];
#pragma unroll
      for (int kk = 0; kk < 4; ++kk) ah[kk] = load_a_frag(&sHC[0][0] + kk*32, 2*HH);
#pragma unroll
      for (int nt = 0; nt < 4; ++nt) {
        int n0 = wave*64 + nt*16;
        v8f acc = {};
        acc = wmma16(ax, load_b_frag(wWih, 32, n0, 0), acc);
#pragma unroll
        for (int kk = 0; kk < 4; ++kk)
          acc = wmma16(ah[kk], load_b_frag(wWhh, HH, n0, kk*32), acc);
        store_d_f16(&sGates[0][0], GG, n0, acc);
      }
    }
    __syncthreads();

    // (4) LSTM epilogue
    for (int idx = tid; idx < MB*HH; idx += NTH) {
      int j = idx & (HH-1), m = idx >> 7;
      float gi = (float)sGates[m][j]        + bias[j];
      float gf = (float)sGates[m][HH+j]     + bias[HH+j];
      float gg = (float)sGates[m][2*HH+j]   + bias[2*HH+j];
      float go = (float)sGates[m][3*HH+j]   + bias[3*HH+j];
      float cn = sig_(gf)*sC[m][j] + sig_(gi)*tanh_(gg);
      float hn = sig_(go)*tanh_(cn);
      sC[m][j]     = cn;
      sHC[m][j]    = (f16)hn;
      sHC[m][HH+j] = (f16)cn;
      encH[(((size_t)(b0+m))*SS + t)*HH + j] = (f16)hn;
      if (t == SS-1) { hEnd[(size_t)(b0+m)*HH + j] = hn; cEnd[(size_t)(b0+m)*HH + j] = cn; }
    }
    __syncthreads();
  }
}

// ---------------- kernel 3: attn_encoder = enc_hidden @ V_d^T (dense WMMA GEMM) ----------------
__global__ __launch_bounds__(NTH)
void attnenc_kernel(char* ws) {
  const f16* encH = (const f16*)(ws + OFF_ENCH);
  const f16* wVd  = (const f16*)(ws + OFF_VD);
  f16* attnE      = (f16*)(ws + OFF_ATTNE);
  size_t row0 = (size_t)blockIdx.x * 16;       // over B*S rows
  int wave = threadIdx.x >> 5;
  int n0 = wave * 16;                          // 8 waves cover N=128
  const f16* A = encH + row0 * HH;
  v8f acc = {};
#pragma unroll
  for (int k0 = 0; k0 < HH; k0 += 32)
    acc = wmma16(load_a_frag(A + k0, HH), load_b_frag(wVd, HH, n0, k0), acc);
  int lane = threadIdx.x & 31;
  int n = n0 + (lane & 15);
  int mb = (lane >> 4) << 3;
#pragma unroll
  for (int i = 0; i < 8; ++i)
    attnE[(row0 + mb + i) * HH + n] = (f16)acc[i];
}

// ---------------- kernel 4: decoder (temporal attention + LSTM + Verlet, 32 steps) -------------
__global__ __launch_bounds__(NTH)
void decoder_kernel(const float* __restrict__ x, const float* __restrict__ vd,
                    const float* __restrict__ Wout, const float* __restrict__ bout,
                    const float* __restrict__ smean, const float* __restrict__ sstd,
                    const float* __restrict__ pmean, float* __restrict__ out, char* ws) {
  __shared__ f16   sDC[MB][2*HH];     //  8 KB [d|c] fp16
  __shared__ float sC[MB][HH];        //  8 KB
  __shared__ float sAttnD[MB][HH];    //  8 KB
  __shared__ float sBeta[MB][SS];     //  4 KB
  __shared__ f16   sGates[MB][GG];    // 16 KB
  __shared__ float sCtx[MB][HH];      //  8 KB
  __shared__ f16   sCtxH[MB][HH];     //  4 KB
  __shared__ f16   sPhys[MB][32];     //  1 KB
  __shared__ float sPos[MB][3], sVel[MB][3], sAcc[MB][3];

  const f16*   wWd   = (const f16*)(ws + OFF_WD);
  const f16*   wWp   = (const f16*)(ws + OFF_WPD);
  const f16*   wWc   = (const f16*)(ws + OFF_WCD);
  const f16*   wWhhD = (const f16*)(ws + OFF_WHHD);
  const float* bias  = (const float*)(ws + OFF_BIASD);
  const f16*   encH  = (const f16*)(ws + OFF_ENCH);
  const f16*   attnE = (const f16*)(ws + OFF_ATTNE);
  const float* hE    = (const float*)(ws + OFF_HENC);
  const float* cE    = (const float*)(ws + OFF_CENC);

  const int b0   = blockIdx.x * MB;
  const int tid  = threadIdx.x;
  const int wave = tid >> 5;

  for (int idx = tid; idx < MB*HH; idx += NTH) {
    int j = idx & (HH-1), m = idx >> 7;
    float h = hE[(size_t)(b0+m)*HH + j], c = cE[(size_t)(b0+m)*HH + j];
    sDC[m][j] = (f16)h; sDC[m][HH+j] = (f16)c; sC[m][j] = c;
  }
  if (tid < MB*3) {
    int a = tid % 3, m = tid / 3;
    const float* xr = x + ((size_t)(b0+m)*SS + (SS-1))*DD;
    const float* xp = x + ((size_t)(b0+m)*SS + (SS-2))*DD;
    float pos  = xr[a]   + pmean[a];
    float vel  = xr[3+a] * sstd[a] + smean[a];
    float pvel = xp[3+a] * sstd[a] + smean[a];
    sPos[m][a] = pos; sVel[m][a] = vel; sAcc[m][a] = (vel - pvel) / DT_;
  }
  __syncthreads();

  for (int z = 0; z < HZN; ++z) {
    // (1) attn_dec = [d|c] @ W_d^T : M=16, N=128, K=256; one N-tile per wave
    {
      int n0 = wave * 16;
      v8f acc = {};
#pragma unroll
      for (int k0 = 0; k0 < 2*HH; k0 += 32)
        acc = wmma16(load_a_frag(&sDC[0][0] + k0, 2*HH),
                     load_b_frag(wWd, 2*HH, n0, k0), acc);
      store_d_f32(&sAttnD[0][0], HH, n0, acc);
    }
    __syncthreads();

    // (2) e[m][s] = sum_h v_d[h]*tanh(attn_d + attn_enc); softmax over S
    for (int idx = tid; idx < MB*SS; idx += NTH) {
      int s = idx % SS, m = idx / SS;
      const f16* ae = attnE + (((size_t)(b0+m))*SS + s)*HH;
      __builtin_prefetch(ae + HH, 0, 0);
      float acc = 0.f;
#pragma unroll 4
      for (int j = 0; j < HH; ++j)
        acc += vd[j] * tanh_(sAttnD[m][j] + (float)ae[j]);
      sBeta[m][s] = acc;
    }
    __syncthreads();
    if (tid < MB) {
      int m = tid; float mx = -1e30f;
      for (int s = 0; s < SS; ++s) mx = fmaxf(mx, sBeta[m][s]);
      float sum = 0.f;
      for (int s = 0; s < SS; ++s) { float e = __expf(sBeta[m][s]-mx); sBeta[m][s] = e; sum += e; }
      float inv = 1.f / sum;
      for (int s = 0; s < SS; ++s) sBeta[m][s] *= inv;
    }
    __syncthreads();

    // (3) context = beta . enc_hidden ; (4) phys vector (padded to K=32)
    for (int idx = tid; idx < MB*HH; idx += NTH) {
      int j = idx & (HH-1), m = idx >> 7;
      const f16* eh = encH + ((size_t)(b0+m))*SS*HH + j;
      float acc = 0.f;
#pragma unroll 4
      for (int s = 0; s < SS; ++s) acc += sBeta[m][s] * (float)eh[s*HH];
      sCtx[m][j] = acc; sCtxH[m][j] = (f16)acc;
    }
    if (tid < MB*32) {
      int k = tid & 31, m = tid >> 5;
      float v = 0.f;
      if (k < 3)      v = sPos[m][k] - pmean[k];
      else if (k < 6) v = (sVel[m][k-3] - smean[k-3]) / sstd[k-3];
      else if (k < 9) v = sAcc[m][k-6] / sstd[k-6];
      sPhys[m][k] = (f16)v;
    }
    __syncthreads();

    // (5) gates = phys@Wp^T + ctx@Wc^T + d@Whh_dec^T : 4 N-tiles/wave, 9 WMMA each
    {
      v16h ap = load_a_frag(&sPhys[0][0], 32);
      v16h ac[4], ad[4];
#pragma unroll
      for (int kk = 0; kk < 4; ++kk) {
        ac[kk] = load_a_frag(&sCtxH[0][0] + kk*32, HH);
        ad[kk] = load_a_frag(&sDC[0][0] + kk*32, 2*HH);
      }
#pragma unroll
      for (int nt = 0; nt < 4; ++nt) {
        int n0 = wave*64 + nt*16;
        v8f acc = {};
        acc = wmma16(ap, load_b_frag(wWp, 32, n0, 0), acc);
#pragma unroll
        for (int kk = 0; kk < 4; ++kk)
          acc = wmma16(ac[kk], load_b_frag(wWc, HH, n0, kk*32), acc);
#pragma unroll
        for (int kk = 0; kk < 4; ++kk)
          acc = wmma16(ad[kk], load_b_frag(wWhhD, HH, n0, kk*32), acc);
        store_d_f16(&sGates[0][0], GG, n0, acc);
      }
    }
    __syncthreads();

    // (6) LSTM epilogue
    for (int idx = tid; idx < MB*HH; idx += NTH) {
      int j = idx & (HH-1), m = idx >> 7;
      float gi = (float)sGates[m][j]        + bias[j];
      float gf = (float)sGates[m][HH+j]     + bias[HH+j];
      float gg = (float)sGates[m][2*HH+j]   + bias[2*HH+j];
      float go = (float)sGates[m][3*HH+j]   + bias[3*HH+j];
      float cn = sig_(gf)*sC[m][j] + sig_(gi)*tanh_(gg);
      float hn = sig_(go)*tanh_(cn);
      sC[m][j] = cn; sDC[m][j] = (f16)hn; sDC[m][HH+j] = (f16)cn;
    }
    __syncthreads();

    // (7) pred_acc = [d|ctx]@W_out^T * std ; Verlet; emit (pos, vel, acc)
    if (tid < MB*3) {
      int a = tid % 3, m = tid / 3;
      float acc = bout[a];
      const float* wr = Wout + a*2*HH;
      for (int j = 0; j < HH; ++j) acc += wr[j]    * (float)sDC[m][j];
      for (int j = 0; j < HH; ++j) acc += wr[HH+j] * sCtx[m][j];
      float pacc = acc * sstd[a];
      float pos = sPos[m][a], vel = sVel[m][a], a0 = sAcc[m][a];
      float posp = pos + vel*DT_ + 0.5f*a0*DT_*DT_;
      float veln = vel + 0.5f*(a0 + pacc)*DT_;
      size_t ob = (((size_t)(b0+m))*HZN + z)*9;
      out[ob + a] = posp; out[ob + 3 + a] = veln; out[ob + 6 + a] = pacc;
      sPos[m][a] = posp; sVel[m][a] = veln; sAcc[m][a] = pacc;
    }
    __syncthreads();
  }
}

// ---------------- launcher ----------------
extern "C" void kernel_launch(void* const* d_in, const int* in_sizes, int n_in,
                              void* d_out, int out_size, void* d_ws, size_t ws_size,
                              hipStream_t stream) {
  (void)in_sizes; (void)n_in; (void)out_size; (void)ws_size;
  const float* x     = (const float*)d_in[0];
  const float* WihE  = (const float*)d_in[1];
  const float* WhhE  = (const float*)d_in[2];
  const float* bihE  = (const float*)d_in[3];
  const float* bhhE  = (const float*)d_in[4];
  const float* We    = (const float*)d_in[5];
  const float* Ve    = (const float*)d_in[6];
  const float* ve    = (const float*)d_in[7];
  const float* Wd    = (const float*)d_in[8];
  const float* Vd    = (const float*)d_in[9];
  const float* vd    = (const float*)d_in[10];
  const float* WihD  = (const float*)d_in[11];
  const float* WhhD  = (const float*)d_in[12];
  const float* bihD  = (const float*)d_in[13];
  const float* bhhD  = (const float*)d_in[14];
  const float* Wout  = (const float*)d_in[15];
  const float* bout  = (const float*)d_in[16];
  const float* smean = (const float*)d_in[17];
  const float* sstd  = (const float*)d_in[18];
  const float* pmean = (const float*)d_in[19];
  char*  ws  = (char*)d_ws;
  float* out = (float*)d_out;

  prep_kernel<<<(PREP_TOTAL + 255)/256, 256, 0, stream>>>(
      WihE, WhhE, bihE, bhhE, We, Wd, Vd, WihD, WhhD, bihD, bhhD, ws);
  encoder_kernel<<<BB/MB, NTH, 0, stream>>>(x, Ve, ve, ws);
  attnenc_kernel<<<(BB*SS)/16, NTH, 0, stream>>>(ws);
  decoder_kernel<<<BB/MB, NTH, 0, stream>>>(x, vd, Wout, bout, smean, sstd, pmean, out, ws);
}
